// TransUPModel_75720273429289
// MI455X (gfx1250) — matrix operand order repaired and checked by reading, not compiled, not run on hardware
//
#include <hip/hip_runtime.h>

// WMMA fragment types (gfx1250, wave32)
typedef __attribute__((ext_vector_type(16))) _Float16 v16h;
typedef __attribute__((ext_vector_type(8)))  _Float16 v8h;
typedef __attribute__((ext_vector_type(4)))  _Float16 v4h;
typedef __attribute__((ext_vector_type(8)))  float    v8f;

#define WAVES_PER_BLOCK 4
#define THREADS (WAVES_PER_BLOCK * 32)
#define DDIM 64
#define PDIM 32

__device__ __forceinline__ v16h cat8(v8h lo, v8h hi) {
  v16h r;
#pragma unroll
  for (int i = 0; i < 8; ++i) { r[i] = lo[i]; r[i + 8] = hi[i]; }
  return r;
}

// Pack sum of two 8-float chunks (as float4 pairs) into 8 f16 lanes of a fragment
__device__ __forceinline__ void put8sum(v16h& dst, int base,
                                        const float4& ua, const float4& ub,
                                        const float4& ia, const float4& ib) {
  dst[base + 0] = (_Float16)(ua.x + ia.x);
  dst[base + 1] = (_Float16)(ua.y + ia.y);
  dst[base + 2] = (_Float16)(ua.z + ia.z);
  dst[base + 3] = (_Float16)(ua.w + ia.w);
  dst[base + 4] = (_Float16)(ub.x + ib.x);
  dst[base + 5] = (_Float16)(ub.y + ib.y);
  dst[base + 6] = (_Float16)(ub.z + ib.z);
  dst[base + 7] = (_Float16)(ub.w + ib.w);
}

__global__ __launch_bounds__(THREADS)
void transup_wmma_kernel(const int* __restrict__ u_ids,
                         const int* __restrict__ i_ids,
                         const float* __restrict__ user_w,
                         const float* __restrict__ item_w,
                         const float* __restrict__ pref_w,
                         const float* __restrict__ norm_w,
                         float* __restrict__ out,
                         int nB, int nTiles)
{
  // Shared constant tables (f16, both orientations as needed by B fragments)
  __shared__ _Float16 prefKD[PDIM * DDIM];  // [p][d] : GEMM1 B (k=d contiguous)
  __shared__ _Float16 prefDK[DDIM * PDIM];  // [d][p] : GEMM2 B (k=p contiguous)
  __shared__ _Float16 normDK[DDIM * PDIM];  // [d][p] : GEMM3 B
  // Per-wave staging (LDS used only where a cross-lane transpose is required)
  __shared__ float    sDif[WAVES_PER_BLOCK][16 * DDIM];  // u-i, f32, row-major
  __shared__ _Float16 sP  [WAVES_PER_BLOCK][16 * PDIM];  // pre_probs, f16

  const int tid  = threadIdx.x;
  const int lane = tid & 31;
  const int wv   = tid >> 5;
  const int hi   = (lane >> 4) & 1;   // lane half (0: lanes 0-15, 1: 16-31)
  const int lc   = lane & 15;

  // ---- stage pref/norm tables into LDS as f16 (vectorized, one-time) ----
  for (int i = tid; i < (PDIM * DDIM) / 4; i += THREADS) {   // 512 float4s
    int p  = i >> 4;            // row (p)
    int d4 = (i & 15) * 4;      // starting d
    float4 fp = ((const float4*)pref_w)[i];
    float4 fn = ((const float4*)norm_w)[i];
    v4h hp; hp[0] = (_Float16)fp.x; hp[1] = (_Float16)fp.y;
            hp[2] = (_Float16)fp.z; hp[3] = (_Float16)fp.w;
    *(v4h*)(&prefKD[p * DDIM + d4]) = hp;
    prefDK[(d4 + 0) * PDIM + p] = hp[0];
    prefDK[(d4 + 1) * PDIM + p] = hp[1];
    prefDK[(d4 + 2) * PDIM + p] = hp[2];
    prefDK[(d4 + 3) * PDIM + p] = hp[3];
    normDK[(d4 + 0) * PDIM + p] = (_Float16)fn.x;
    normDK[(d4 + 1) * PDIM + p] = (_Float16)fn.y;
    normDK[(d4 + 2) * PDIM + p] = (_Float16)fn.z;
    normDK[(d4 + 3) * PDIM + p] = (_Float16)fn.w;
  }

  int tile = blockIdx.x * WAVES_PER_BLOCK + wv;
  const bool valid = (tile < nTiles);
  if (tile >= nTiles) tile = nTiles - 1;       // clamp; keep EXEC uniform for WMMA
  const int rowBase = tile * 16;

  // ---- gather: lane (lc,hi) owns row lc; the two halves split the 256B row
  //      into the exact 8-float chunks the WMMA A-layout wants ----
  int bidx = rowBase + lc; if (bidx >= nB) bidx = nB - 1;
  const float* uRow = user_w + (long long)u_ids[bidx] * DDIM;
  const float* iRow = item_w + (long long)i_ids[bidx] * DDIM;

  float4 u4[8], i4[8];
#pragma unroll
  for (int c = 0; c < 4; ++c) {
    int off = c * 16 + hi * 8;                 // float offset of this 8-float chunk
    u4[2*c]     = *(const float4*)(uRow + off);
    u4[2*c + 1] = *(const float4*)(uRow + off + 4);
    i4[2*c]     = *(const float4*)(iRow + off);
    i4[2*c + 1] = *(const float4*)(iRow + off + 4);
  }

  // d = u - i staged to LDS (needed later in the transposed C/D layout)
#pragma unroll
  for (int c = 0; c < 4; ++c) {
    int off = c * 16 + hi * 8;
    float4 a = u4[2*c],     b = i4[2*c];
    float4 d0; d0.x = a.x - b.x; d0.y = a.y - b.y; d0.z = a.z - b.z; d0.w = a.w - b.w;
    a = u4[2*c + 1]; b = i4[2*c + 1];
    float4 d1; d1.x = a.x - b.x; d1.y = a.y - b.y; d1.z = a.z - b.z; d1.w = a.w - b.w;
    *(float4*)(&sDif[wv][lc * DDIM + off])     = d0;
    *(float4*)(&sDif[wv][lc * DDIM + off + 4]) = d1;
  }

  // A fragments (u+i as f16) built directly in registers — no LDS round-trip.
  // a0 (K=0..31): e<8 -> K = hi*8+e (chunk c=0); e>=8 -> K = 16+hi*8+(e-8) (c=1)
  v16h a0, a1;
  put8sum(a0, 0, u4[0], u4[1], i4[0], i4[1]);
  put8sum(a0, 8, u4[2], u4[3], i4[2], i4[3]);
  put8sum(a1, 0, u4[4], u4[5], i4[4], i4[5]);
  put8sum(a1, 8, u4[6], u4[7], i4[6], i4[7]);

  __syncthreads();

  // ---- GEMM1: S = 0.5 * (u+i) @ pref^T  [16x64]x[64x32] -> [16x32] ----
  v8f s0 = {}, s1 = {};
  {
    const _Float16* bp = prefKD + lc * DDIM;             // n = p = lc
    v16h b00 = cat8(*(const v8h*)(bp + hi * 8),      *(const v8h*)(bp + 16 + hi * 8));
    v16h b01 = cat8(*(const v8h*)(bp + 32 + hi * 8), *(const v8h*)(bp + 48 + hi * 8));
    s0 = __builtin_amdgcn_wmma_f32_16x16x32_f16(false, a0, false, b00, (short)0, s0, false, false);
    s0 = __builtin_amdgcn_wmma_f32_16x16x32_f16(false, a1, false, b01, (short)0, s0, false, false);
  }
  {
    const _Float16* bp = prefKD + (16 + lc) * DDIM;      // n = p = 16+lc
    v16h b10 = cat8(*(const v8h*)(bp + hi * 8),      *(const v8h*)(bp + 16 + hi * 8));
    v16h b11 = cat8(*(const v8h*)(bp + 32 + hi * 8), *(const v8h*)(bp + 48 + hi * 8));
    s1 = __builtin_amdgcn_wmma_f32_16x16x32_f16(false, a0, false, b10, (short)0, s1, false, false);
    s1 = __builtin_amdgcn_wmma_f32_16x16x32_f16(false, a1, false, b11, (short)0, s1, false, false);
  }

  // scale by 0.5 and restage as f16 A operand for GEMM2/3 (cross-lane transpose)
#pragma unroll
  for (int v = 0; v < 8; ++v) {
    int m = v + hi * 8;
    sP[wv][m * PDIM + lc]      = (_Float16)(0.5f * s0[v]);
    sP[wv][m * PDIM + 16 + lc] = (_Float16)(0.5f * s1[v]);
  }
  __syncthreads();

  // ---- GEMM2/3: R = S @ pref, N = S @ norm  [16x32]x[32x64] -> [16x64] ----
  const _Float16* myP = &sP[wv][0];
  v16h ap = cat8(*(const v8h*)(myP + lc * PDIM + hi * 16),
                 *(const v8h*)(myP + lc * PDIM + hi * 16 + 8));

  v8f R[4], Nm[4];
#pragma unroll
  for (int nt = 0; nt < 4; ++nt) {
    const _Float16* bp = prefDK + (nt * 16 + lc) * PDIM + hi * 16;  // n=d, k=p contig
    v16h br = cat8(*(const v8h*)bp, *(const v8h*)(bp + 8));
    v8f z0 = {};
    R[nt] = __builtin_amdgcn_wmma_f32_16x16x32_f16(false, ap, false, br, (short)0, z0, false, false);
    const _Float16* bn = normDK + (nt * 16 + lc) * PDIM + hi * 16;
    v16h bb = cat8(*(const v8h*)bn, *(const v8h*)(bn + 8));
    v8f z1 = {};
    Nm[nt] = __builtin_amdgcn_wmma_f32_16x16x32_f16(false, ap, false, bb, (short)0, z1, false, false);
  }

  // ---- elementwise: out = sum_d | d + r - (d.n) * n |,  d = u-i (f32) ----
  // Read d in C/D layout: lane -> column D = nt*16+lc, VGPR v -> row M = v+8*hi
  const float* myD = &sDif[wv][0];
  float dv[4][8];
  float tpart[8];
#pragma unroll
  for (int v = 0; v < 8; ++v) tpart[v] = 0.0f;
#pragma unroll
  for (int nt = 0; nt < 4; ++nt) {
#pragma unroll
    for (int v = 0; v < 8; ++v) {
      int m = v + hi * 8;
      float dd = myD[m * DDIM + nt * 16 + lc];
      dv[nt][v] = dd;
      tpart[v] += dd * Nm[nt][v];
    }
  }
  // reduce d.n across the 16 lanes of each half (xor masks stay within half)
#pragma unroll
  for (int v = 0; v < 8; ++v) {
#pragma unroll
    for (int off = 1; off < 16; off <<= 1)
      tpart[v] += __shfl_xor(tpart[v], off, 32);
  }

  float acc[8];
#pragma unroll
  for (int v = 0; v < 8; ++v) {
    float a = 0.0f;
#pragma unroll
    for (int nt = 0; nt < 4; ++nt)
      a += fabsf(dv[nt][v] + R[nt][v] - tpart[v] * Nm[nt][v]);
    acc[v] = a;
  }
#pragma unroll
  for (int v = 0; v < 8; ++v) {
#pragma unroll
    for (int off = 1; off < 16; off <<= 1)
      acc[v] += __shfl_xor(acc[v], off, 32);
  }

  // lane 0 of each half writes its 8 rows (lane 0 -> rows 0..7, lane 16 -> 8..15)
  if (valid && lc == 0) {
#pragma unroll
    for (int v = 0; v < 8; ++v) {
      int row = rowBase + hi * 8 + v;
      if (row < nB) out[row] = acc[v];
    }
  }
}

extern "C" void kernel_launch(void* const* d_in, const int* in_sizes, int n_in,
                              void* d_out, int out_size, void* d_ws, size_t ws_size,
                              hipStream_t stream) {
  (void)n_in; (void)out_size; (void)d_ws; (void)ws_size;
  const int*   u_ids  = (const int*)d_in[0];
  const int*   i_ids  = (const int*)d_in[1];
  const float* user_w = (const float*)d_in[2];
  const float* item_w = (const float*)d_in[3];
  const float* pref_w = (const float*)d_in[4];
  const float* norm_w = (const float*)d_in[5];
  float* out = (float*)d_out;

  const int nB     = in_sizes[0];
  const int nTiles = (nB + 15) / 16;
  const int blocks = (nTiles + WAVES_PER_BLOCK - 1) / WAVES_PER_BLOCK;

  transup_wmma_kernel<<<blocks, THREADS, 0, stream>>>(
      u_ids, i_ids, user_w, item_w, pref_w, norm_w, out, nB, nTiles);
}
